// ComENet_24163486008144
// MI455X (gfx1250) — matrix operand order repaired and checked by reading, not compiled, not run on hardware
//
#include <hip/hip_runtime.h>
#include <hip/hip_bf16.h>

// ---------------------------------------------------------------------------
// ComENet block forward on MI455X (gfx1250), bf16 WMMA everywhere.
// ---------------------------------------------------------------------------

typedef __bf16 bf16;
typedef __attribute__((ext_vector_type(16))) __bf16 v16bf;
typedef __attribute__((ext_vector_type(8)))  __bf16 v8bf;
typedef __attribute__((ext_vector_type(8)))  float  v8f;

#define NNODES 50000
#define NEDGES 1000000
#define HID    128
#define NGRAPH 512

__device__ inline v8f zero8() {
  v8f z;
#pragma unroll
  for (int i = 0; i < 8; ++i) z[i] = 0.f;
  return z;
}

__device__ inline v8f wmma_bf16(v16bf a, v16bf b, v8f c) {
  // D = A(16x32 bf16) * B(32x16 bf16) + C(16x16 f32)
  return __builtin_amdgcn_wmma_f32_16x16x32_bf16(
      /*neg_a=*/false, a, /*neg_b=*/false, b,
      /*c_mod=*/(short)0, c, /*reuse_a=*/false, /*reuse_b=*/false);
}

// A-fragment loader: lane (half = lane>>4, m = lane&15) holds
//   elements 0..7  = A[m][k0 + half*16 + 0..7]
//   elements 8..15 = A[m][k0 + half*16 + 8 + 0..7]
// `row` points at Alds[m][0]; all LDS strides are chosen so these two
// 16-byte loads are 16B-aligned (ds_load_b128).
__device__ inline v16bf load_a_frag(const bf16* row, int k0, int half) {
  v8bf lo = *(const v8bf*)(row + k0 + half * 16);
  v8bf hi = *(const v8bf*)(row + k0 + half * 16 + 8);
  v16bf a;
#pragma unroll
  for (int i = 0; i < 8; ++i) { a[i] = lo[i]; a[i + 8] = hi[i]; }
  return a;
}

// ---------------------------------------------------------------------------
// Weight packer: W[k][n] (f32, row-major [Kin, Nout]) -> B-fragment layout:
//   Wp[((g)*Nout + n)*16 + e] = bf16(W[g*16 + e][n]),  zero if k >= Kin.
// A lane's B fragment is then 16 contiguous bf16 (32B).
// ---------------------------------------------------------------------------
__global__ void k_pack(const float* __restrict__ W, bf16* __restrict__ Wp,
                       int Kin, int Kpad, int Nout) {
  int t = blockIdx.x * blockDim.x + threadIdx.x;
  int total = Kpad * Nout;
  if (t >= total) return;
  int e = t & 15;
  int rest = t >> 4;
  int n = rest % Nout;
  int g = rest / Nout;
  int k = g * 16 + e;
  Wp[t] = (k < Kin) ? (bf16)W[(size_t)k * Nout + n] : (bf16)0.f;
}

__global__ void k_zero(float* __restrict__ p, int n) {
  int i = blockIdx.x * blockDim.x + threadIdx.x;
  if (i < n) p[i] = 0.f;
}

// ---------------------------------------------------------------------------
// Generic node GEMM:  out[r, 0:128] = epi( A[r, 0:Kin] @ W + bias )
//   A = A0 (or concat(A0, A1) along features when A1 != null, Kin == 256)
//   epi: x = acc + bias; if(act) x = swish(x); if(resid) x += resid; store.
// Block = 256 threads (8 waves), 16 rows per block, each wave owns 16 cols.
// ---------------------------------------------------------------------------
__global__ __launch_bounds__(256)
void k_gemm(const float* __restrict__ A0, const float* __restrict__ A1,
            const bf16* __restrict__ Wp, const float* __restrict__ bias,
            const float* __restrict__ resid, float* __restrict__ out,
            int nrows, int Kin, int act) {
  __shared__ bf16 Alds[16 * 264];          // supports Kin up to 256
  const int AST = (Kin > 128) ? 264 : 136; // stride ≡ 8 (mod 16) elems -> 16B-aligned rows
  const int tid = threadIdx.x;
  const int rowbase = blockIdx.x * 16;
  const int s0 = A1 ? 128 : Kin;

  for (int i = tid; i < 16 * Kin; i += 256) {
    int r = i / Kin, c = i - r * Kin;
    int row = rowbase + r;
    float v = 0.f;
    if (row < nrows) {
      if (A1 && c >= 128) v = A1[(size_t)row * 128 + (c - 128)];
      else                v = A0[(size_t)row * s0 + c];
    }
    Alds[r * AST + c] = (bf16)v;
  }
  __syncthreads();

  const int wave = tid >> 5, lane = tid & 31;
  const int half = lane >> 4, m = lane & 15;
  const int col = wave * 16 + (lane & 15);

  v8f acc = zero8();
  const int nkb = Kin >> 5;
  for (int kb = 0; kb < nkb; ++kb) {
    v16bf a = load_a_frag(&Alds[m * AST], kb * 32, half);
    v16bf b = *(const v16bf*)(Wp + (((size_t)(kb * 2 + half)) * 128 + col) * 16);
    acc = wmma_bf16(a, b, acc);
  }

  const float bv = bias ? bias[col] : 0.f;
#pragma unroll
  for (int q = 0; q < 8; ++q) {
    int row = rowbase + half * 8 + q;
    if (row < nrows) {
      float x = acc[q] + bv;
      if (act) x = x / (1.f + __expf(-x));
      if (resid) x += resid[(size_t)row * 128 + col];
      out[(size_t)row * 128 + col] = x;
    }
  }
}

// ---------------------------------------------------------------------------
// Fused edge kernel, 16 edges per block:
//   f1 = (feat1 @ W1a) @ W1b ; f2 = (feat2 @ W2a) @ W2b   (bf16 WMMA)
//   agg1[dst] += f1 * xh[src] ; agg2[dst] += f2 * xh[src] (f32 atomics, L2)
// K padded: 147->160, 21->32. Middle 16x64 tiles staged in LDS as bf16.
// ---------------------------------------------------------------------------
__global__ __launch_bounds__(256)
void k_edges(const float* __restrict__ feat1, const float* __restrict__ feat2,
             const int* __restrict__ eidx, const float* __restrict__ xh,
             const bf16* __restrict__ w1a, const bf16* __restrict__ w1b,
             const bf16* __restrict__ w2a, const bf16* __restrict__ w2b,
             float* __restrict__ agg1, float* __restrict__ agg2, int E_) {
  __shared__ bf16 s1[16 * 168];  // feat1 tile, K padded to 160
  __shared__ bf16 s2[16 * 40];   // feat2 tile, K padded to 32
  __shared__ bf16 t1[16 * 72];   // middle 16x64 (f1 path)
  __shared__ bf16 t2[16 * 72];   // middle 16x64 (f2 path)
  __shared__ int ssrc[16], sdst[16];

  const int tid = threadIdx.x;
  const long ebase = (long)blockIdx.x * 16;

  if (tid < 16) {
    long e = ebase + tid;
    int s_ = 0, d_ = 0;
    if (e < E_) { s_ = eidx[e]; d_ = eidx[(long)E_ + e]; }
    ssrc[tid] = s_; sdst[tid] = d_;
  }
  for (int i = tid; i < 16 * 160; i += 256) {
    int r = i / 160, c = i - r * 160;
    long e = ebase + r;
    float v = (c < 147 && e < E_) ? feat1[e * 147L + c] : 0.f;
    s1[r * 168 + c] = (bf16)v;
  }
  for (int i = tid; i < 16 * 32; i += 256) {
    int r = i >> 5, c = i & 31;
    long e = ebase + r;
    float v = (c < 21 && e < E_) ? feat2[e * 21L + c] : 0.f;
    s2[r * 40 + c] = (bf16)v;
  }
  __syncthreads();

  const int wave = tid >> 5, lane = tid & 31;
  const int half = lane >> 4, m = lane & 15, n = lane & 15;

  // ---- stage 1: waves 0..3 -> t1 (K=160), waves 4..7 -> t2 (K=32) ----
  if (wave < 4) {
    int col = wave * 16 + n;
    v8f acc = zero8();
    for (int kb = 0; kb < 5; ++kb) {
      v16bf a = load_a_frag(&s1[m * 168], kb * 32, half);
      v16bf b = *(const v16bf*)(w1a + (((size_t)(kb * 2 + half)) * 64 + col) * 16);
      acc = wmma_bf16(a, b, acc);
    }
#pragma unroll
    for (int q = 0; q < 8; ++q) t1[(half * 8 + q) * 72 + col] = (bf16)acc[q];
  } else {
    int col = (wave - 4) * 16 + n;
    v8f acc = zero8();
    v16bf a = load_a_frag(&s2[m * 40], 0, half);
    v16bf b = *(const v16bf*)(w2a + ((size_t)half * 64 + col) * 16);
    acc = wmma_bf16(a, b, acc);
#pragma unroll
    for (int q = 0; q < 8; ++q) t2[(half * 8 + q) * 72 + col] = (bf16)acc[q];
  }
  __syncthreads();

  // ---- stage 2: every wave does its 16 cols of both f1 and f2 (K=64) ----
  const int col = wave * 16 + n;
  v8f acc1 = zero8(), acc2 = zero8();
  for (int kb = 0; kb < 2; ++kb) {
    v16bf a = load_a_frag(&t1[m * 72], kb * 32, half);
    v16bf b = *(const v16bf*)(w1b + (((size_t)(kb * 2 + half)) * 128 + col) * 16);
    acc1 = wmma_bf16(a, b, acc1);
  }
  for (int kb = 0; kb < 2; ++kb) {
    v16bf a = load_a_frag(&t2[m * 72], kb * 32, half);
    v16bf b = *(const v16bf*)(w2b + (((size_t)(kb * 2 + half)) * 128 + col) * 16);
    acc2 = wmma_bf16(a, b, acc2);
  }

  // ---- epilogue: gather xh[src] once, scatter into both aggregates ----
#pragma unroll
  for (int q = 0; q < 8; ++q) {
    int mm = half * 8 + q;
    long e = ebase + mm;
    if (e < E_) {
      int s_ = ssrc[mm], d_ = sdst[mm];
      float xv = xh[(size_t)s_ * 128 + col];
      atomicAdd(&agg1[(size_t)d_ * 128 + col], acc1[q] * xv);
      atomicAdd(&agg2[(size_t)d_ * 128 + col], acc2[q] * xv);
    }
  }
}

// --------------------------- GraphNorm kernels -----------------------------
__global__ void k_gn_sum(const float* __restrict__ h, const int* __restrict__ batch,
                         float* __restrict__ gsum, float* __restrict__ gcnt, int total) {
  int i = blockIdx.x * blockDim.x + threadIdx.x;
  if (i >= total) return;
  int row = i >> 7, c = i & 127;
  int g = batch[row];
  atomicAdd(&gsum[g * 128 + c], h[i]);
  if (c == 0) atomicAdd(&gcnt[g], 1.f);
}

__global__ void k_gn_center(float* __restrict__ h, const int* __restrict__ batch,
                            const float* __restrict__ gsum, const float* __restrict__ gcnt,
                            const float* __restrict__ ms, float* __restrict__ gvar, int total) {
  int i = blockIdx.x * blockDim.x + threadIdx.x;
  if (i >= total) return;
  int row = i >> 7, c = i & 127;
  int g = batch[row];
  float cnt = fmaxf(gcnt[g], 1.f);
  float mean = gsum[g * 128 + c] / cnt;
  float o = h[i] - mean * ms[c];
  h[i] = o;
  atomicAdd(&gvar[g * 128 + c], o * o);
}

__global__ void k_gn_final(float* __restrict__ h, const int* __restrict__ batch,
                           const float* __restrict__ gvar, const float* __restrict__ gcnt,
                           const float* __restrict__ w, const float* __restrict__ b, int total) {
  int i = blockIdx.x * blockDim.x + threadIdx.x;
  if (i >= total) return;
  int row = i >> 7, c = i & 127;
  int g = batch[row];
  float cnt = fmaxf(gcnt[g], 1.f);
  float inv = rsqrtf(gvar[g * 128 + c] / cnt + 1e-5f);
  h[i] = w[c] * h[i] * inv + b[c];
}

// ---------------------------------------------------------------------------
extern "C" void kernel_launch(void* const* d_in, const int* in_sizes, int n_in,
                              void* d_out, int out_size, void* d_ws, size_t ws_size,
                              hipStream_t stream) {
  (void)in_sizes; (void)n_in; (void)out_size; (void)ws_size;

  const float* x        = (const float*)d_in[0];
  const float* feature1 = (const float*)d_in[1];
  const float* feature2 = (const float*)d_in[2];
  const int*   edge_idx = (const int*)d_in[3];
  const int*   batch    = (const int*)d_in[4];
  const float* lin_w = (const float*)d_in[5];   const float* lin_b = (const float*)d_in[6];
  const float* f1_w1 = (const float*)d_in[7];   const float* f1_w2 = (const float*)d_in[8];
  const float* f2_w1 = (const float*)d_in[9];   const float* f2_w2 = (const float*)d_in[10];
  const float* c1_rel_w  = (const float*)d_in[11]; const float* c1_rel_b = (const float*)d_in[12];
  const float* c1_root_w = (const float*)d_in[13];
  const float* c2_rel_w  = (const float*)d_in[14]; const float* c2_rel_b = (const float*)d_in[15];
  const float* c2_root_w = (const float*)d_in[16];
  const float* lin1_w = (const float*)d_in[17]; const float* lin1_b = (const float*)d_in[18];
  const float* lin2_w = (const float*)d_in[19]; const float* lin2_b = (const float*)d_in[20];
  const float* lincat_w = (const float*)d_in[21]; const float* lincat_b = (const float*)d_in[22];
  const float* norm_w = (const float*)d_in[23]; const float* norm_b = (const float*)d_in[24];
  const float* norm_ms = (const float*)d_in[25];
  const float* lins_w = (const float*)d_in[26]; const float* lins_b = (const float*)d_in[27];
  const float* final_w = (const float*)d_in[28]; const float* final_b = (const float*)d_in[29];
  float* out = (float*)d_out;

  // ---------------- workspace layout ----------------
  char* ws = (char*)d_ws;
  bf16* pw = (bf16*)ws;                            // packed-weight arena (<1MB)
  const size_t K128 = 128 * 128;                   // bf16 elems per 128x128
  bf16* p_lin    = pw;
  bf16* p_c1root = p_lin + K128;
  bf16* p_c1rel  = p_c1root + K128;
  bf16* p_c2root = p_c1rel + K128;
  bf16* p_c2rel  = p_c2root + K128;
  bf16* p_lin1   = p_c2rel + K128;
  bf16* p_lin2   = p_lin1 + K128;
  bf16* p_lincat = p_lin2 + K128;                  // 256x128
  bf16* p_lins0  = p_lincat + 256 * 128;
  bf16* p_lins1  = p_lins0 + K128;
  bf16* p_lins2  = p_lins1 + K128;
  bf16* p_final  = p_lins2 + K128;
  bf16* p_f1a    = p_final + K128;                 // 160x64
  bf16* p_f1b    = p_f1a + 160 * 64;               // 64x128
  bf16* p_f2a    = p_f1b + 64 * 128;               // 32x64
  bf16* p_f2b    = p_f2a + 32 * 64;                // 64x128

  const size_t NB = (size_t)NNODES * HID * sizeof(float); // 25.6 MB
  float* xh   = (float*)(ws + (1 << 20));
  float* agg1 = (float*)((char*)xh + NB);
  float* agg2 = (float*)((char*)agg1 + NB);
  float* t1   = (float*)((char*)agg2 + NB);
  float* t2   = (float*)((char*)t1 + NB);
  float* h    = (float*)((char*)t2 + NB);
  float* gsum = (float*)((char*)h + NB);                   // G*128
  float* gvar = gsum + NGRAPH * HID;
  float* gcnt = gvar + NGRAPH * HID;                       // G

  // ---------------- weight packing ----------------
  auto pack = [&](const float* W, bf16* Wp, int Kin, int Kpad, int Nout) {
    int tot = Kpad * Nout;
    k_pack<<<(tot + 255) / 256, 256, 0, stream>>>(W, Wp, Kin, Kpad, Nout);
  };
  pack(lin_w,     p_lin,    128, 128, 128);
  pack(c1_root_w, p_c1root, 128, 128, 128);
  pack(c1_rel_w,  p_c1rel,  128, 128, 128);
  pack(c2_root_w, p_c2root, 128, 128, 128);
  pack(c2_rel_w,  p_c2rel,  128, 128, 128);
  pack(lin1_w,    p_lin1,   128, 128, 128);
  pack(lin2_w,    p_lin2,   128, 128, 128);
  pack(lincat_w,  p_lincat, 256, 256, 128);
  pack(lins_w + 0 * 128 * 128, p_lins0, 128, 128, 128);
  pack(lins_w + 1 * 128 * 128, p_lins1, 128, 128, 128);
  pack(lins_w + 2 * 128 * 128, p_lins2, 128, 128, 128);
  pack(final_w, p_final, 128, 128, 128);
  pack(f1_w1, p_f1a, 147, 160, 64);
  pack(f1_w2, p_f1b,  64,  64, 128);
  pack(f2_w1, p_f2a,  21,  32, 64);
  pack(f2_w2, p_f2b,  64,  64, 128);

  // zero aggregates (contiguous) and graphnorm accumulators (contiguous)
  {
    int n1 = 2 * NNODES * HID;
    k_zero<<<(n1 + 255) / 256, 256, 0, stream>>>(agg1, n1);
    int n2 = 2 * NGRAPH * HID + NGRAPH;
    k_zero<<<(n2 + 255) / 256, 256, 0, stream>>>(gsum, n2);
  }

  const int gN = (NNODES + 15) / 16;   // 3125 row-tiles
  const int gE = (NEDGES + 15) / 16;   // 62500 edge-tiles

  // xh = swish(x @ lin_w + lin_b)
  k_gemm<<<gN, 256, 0, stream>>>(x, nullptr, p_lin, lin_b, nullptr, xh, NNODES, 128, 1);

  // fused edge MLPs + gather/scatter into agg1/agg2
  k_edges<<<gE, 256, 0, stream>>>(feature1, feature2, edge_idx, xh,
                                  p_f1a, p_f1b, p_f2a, p_f2b, agg1, agg2, NEDGES);

  // conv1 chain: t1 = xh@root; t1 = agg1@rel + b + t1; t1 = swish(t1@lin1 + b)
  k_gemm<<<gN, 256, 0, stream>>>(xh, nullptr, p_c1root, nullptr, nullptr, t1, NNODES, 128, 0);
  k_gemm<<<gN, 256, 0, stream>>>(agg1, nullptr, p_c1rel, c1_rel_b, t1, t1, NNODES, 128, 0);
  k_gemm<<<gN, 256, 0, stream>>>(t1, nullptr, p_lin1, lin1_b, nullptr, t1, NNODES, 128, 1);

  // conv2 chain
  k_gemm<<<gN, 256, 0, stream>>>(xh, nullptr, p_c2root, nullptr, nullptr, t2, NNODES, 128, 0);
  k_gemm<<<gN, 256, 0, stream>>>(agg2, nullptr, p_c2rel, c2_rel_b, t2, t2, NNODES, 128, 0);
  k_gemm<<<gN, 256, 0, stream>>>(t2, nullptr, p_lin2, lin2_b, nullptr, t2, NNODES, 128, 1);

  // h = concat(h1,h2) @ lincat + b + xh
  k_gemm<<<gN, 256, 0, stream>>>(t1, t2, p_lincat, lincat_b, xh, h, NNODES, 256, 0);

  // residual stack: h = swish(h@W_i + b_i) + h
  k_gemm<<<gN, 256, 0, stream>>>(h, nullptr, p_lins0, lins_b + 0 * 128, h, h, NNODES, 128, 1);
  k_gemm<<<gN, 256, 0, stream>>>(h, nullptr, p_lins1, lins_b + 1 * 128, h, h, NNODES, 128, 1);
  k_gemm<<<gN, 256, 0, stream>>>(h, nullptr, p_lins2, lins_b + 2 * 128, h, h, NNODES, 128, 1);

  // GraphNorm
  {
    int tot = NNODES * HID;
    int g = (tot + 255) / 256;
    k_gn_sum<<<g, 256, 0, stream>>>(h, batch, gsum, gcnt, tot);
    k_gn_center<<<g, 256, 0, stream>>>(h, batch, gsum, gcnt, norm_ms, gvar, tot);
    k_gn_final<<<g, 256, 0, stream>>>(h, batch, gvar, gcnt, norm_w, norm_b, tot);
  }

  // out = h @ final_w + final_b
  k_gemm<<<gN, 256, 0, stream>>>(h, nullptr, p_final, final_b, nullptr, out, NNODES, 128, 0);
}
